// GenNet_63410897158782
// MI455X (gfx1250) — matrix operand order repaired and checked by reading, not compiled, not run on hardware
//
#include <hip/hip_runtime.h>

// ---------------------------------------------------------------------------
// Autoregressive 2-layer LSTM (B=256, H=300, G=1200, T=1024) for gfx1250.
//
// Persistent kernel, 152 blocks x 256 threads. Key structure:
//  * Weight rows repacked as g' = gate*304 + h (G padded to 1216) so one block
//    owns all 4 gates of a 16-wide h-slice for 2 b-tiles (8 waves, 8 tiles).
//  * Weight WMMA fragments live in VGPRs for the whole 1024-step loop.
//  * LSTM batch rows never mix -> the grid is 8 *independent* groups of 19
//    blocks (one per 32-row batch slice); each group has its own 19-member
//    atomic barrier (2 per step, the minimum for the recurrence).
//  * Per phase, the block pulls the 40KB A-slab (both A matrices, 32 rows x
//    320 cols bf16) from L2 straight into LDS with CDNA5 async copies
//    (global_load_async_to_lds_b128, ASYNCcnt), then feeds WMMA from cheap
//    ds_load_b128 fragments.
//  * Gates are exchanged through LDS; c-state lives in LDS permanently.
// ---------------------------------------------------------------------------

#define Bb   256
#define Hh   300
#define Hp   320          // K dimension padded to a multiple of 32
#define HT   304          // per-gate column pad (19 tiles of 16)
#define Gp   1216         // 4 * HT
#define Tt   1024
#define TAGD 384
#define NGT  19           // h-tiles == blocks per batch group
#define NWG  (8 * NGT)    // 152 blocks
#define NTHR 256
#define KC   (Hp / 32)    // 10 K-chunks

typedef __bf16 v16bf __attribute__((ext_vector_type(16)));
typedef float  v8f   __attribute__((ext_vector_type(8)));

union FragBF {
    uint4 q[2];
    v16bf v;
};

__device__ __forceinline__ unsigned short f2bf(float f) {
    unsigned u = __float_as_uint(f);
    unsigned r = u + 0x7FFFu + ((u >> 16) & 1u);   // round-to-nearest-even
    return (unsigned short)(r >> 16);
}

__device__ __forceinline__ float sigmoidf_(float x) {
    return 1.0f / (1.0f + __expf(-x));
}

// B fragment (contiguous k order, same as A): lane&15 = column, lane half selects 16
// contiguous K values -> two adjacent 16B loads.
__device__ __forceinline__ void load_bfrag(FragBF& f, const unsigned short* p,
                                           int off) {
    f.q[0] = *(const uint4*)(p + off);
    f.q[1] = *(const uint4*)(p + off + 8);
}

// --------------------- per-batch-group barrier (19 blocks) ------------------
__device__ __forceinline__ void group_barrier(unsigned* cnt, unsigned* gen) {
    __syncthreads();
    if (threadIdx.x == 0) {
        __threadfence();
        unsigned g = __hip_atomic_load(gen, __ATOMIC_RELAXED,
                                       __HIP_MEMORY_SCOPE_AGENT);
        unsigned t = __hip_atomic_fetch_add(cnt, 1u, __ATOMIC_ACQ_REL,
                                            __HIP_MEMORY_SCOPE_AGENT);
        if (t == NGT - 1) {
            __hip_atomic_store(cnt, 0u, __ATOMIC_RELAXED,
                               __HIP_MEMORY_SCOPE_AGENT);
            __hip_atomic_fetch_add(gen, 1u, __ATOMIC_ACQ_REL,
                                   __HIP_MEMORY_SCOPE_AGENT);
        } else {
            while (__hip_atomic_load(gen, __ATOMIC_ACQUIRE,
                                     __HIP_MEMORY_SCOPE_AGENT) == g) {
                __builtin_amdgcn_s_sleep(1);
            }
        }
        __threadfence();
    }
    __syncthreads();
}

// ----------------------------- kernels --------------------------------------
__global__ void init_state_kernel(
    unsigned* bar,
    unsigned short* bx, unsigned short* bh0, unsigned short* bh1,
    const float* __restrict__ bih0, const float* __restrict__ bhh0,
    const float* __restrict__ bih1, const float* __restrict__ bhh1,
    float* biasP0, float* biasP1)
{
    const int i = blockIdx.x * blockDim.x + threadIdx.x;
    if (i < 8 * 64) bar[i] = 0u;                 // 8 group barriers
    if (i < Bb * Hp) { bx[i] = 0; bh0[i] = 0; bh1[i] = 0; }
    if (i < Gp) {
        const int gate = i / HT;
        const int hcol = i - gate * HT;
        if (hcol < Hh) {
            const int g = gate * Hh + hcol;
            biasP0[i] = bih0[g] + bhh0[g];
            biasP1[i] = bih1[g] + bhh1[g];
        } else {
            biasP0[i] = 0.0f;
            biasP1[i] = 0.0f;
        }
    }
}

// Repack weights: row g' = gate*HT + hcol, K padded to Hp, bf16.
__global__ void conv_w_kernel(const float* __restrict__ w,
                              unsigned short* __restrict__ wb)
{
    const int i = blockIdx.x * blockDim.x + threadIdx.x;
    if (i >= Gp * Hp) return;
    const int gp = i / Hp;
    const int k  = i - gp * Hp;
    const int gate = gp / HT;
    const int hcol = gp - gate * HT;
    unsigned short v = 0;
    if (hcol < Hh && k < Hh)
        v = f2bf(w[(size_t)(gate * Hh + hcol) * Hh + k]);
    wb[i] = v;
}

__global__ void fc_proj_kernel(const float* __restrict__ tag,
                               const float* __restrict__ fw,
                               const float* __restrict__ fb,
                               unsigned short* __restrict__ bx)
{
    const int i = blockIdx.x * blockDim.x + threadIdx.x;
    if (i >= Bb * Hh) return;
    const int b = i / Hh;
    const int h = i - b * Hh;
    const float* tv = tag + (size_t)b * TAGD;
    const float* wr = fw + (size_t)h * TAGD;
    float s = fb[h];
    for (int k = 0; k < TAGD; ++k) s = fmaf(tv[k], wr[k], s);
    bx[(size_t)b * Hp + h] = f2bf(s);
}

__global__ void __launch_bounds__(NTHR)
lstm_persist_kernel(
    unsigned short* __restrict__ bx,    // layer0 input  [256][Hp] bf16
    unsigned short* __restrict__ bh0,   // layer0 hidden [256][Hp] bf16
    unsigned short* __restrict__ bh1,   // layer1 hidden [256][Hp] bf16
    const unsigned short* __restrict__ Wih0,   // [Gp][Hp] bf16 (repacked)
    const unsigned short* __restrict__ Whh0,
    const unsigned short* __restrict__ Wih1,
    const unsigned short* __restrict__ Whh1,
    const float* __restrict__ biasP0,   // [Gp]
    const float* __restrict__ biasP1,
    float* __restrict__ out,            // [256][T][300] f32
    unsigned* __restrict__ bar)         // 8 groups x 64 words
{
    __shared__ __align__(16) unsigned short a_s[2][32][Hp];  // 40KB A slab
    __shared__ float gates_s[2][4][16][16];                  // 8KB
    __shared__ float c_s[2][2][16][16];                      // 4KB

    const int wave = threadIdx.x >> 5;
    const int lane = threadIdx.x & 31;
    const int gate = wave & 3;               // this wave's gate tile
    const int bsel = wave >> 2;              // which of the block's 2 b-tiles
    const int bp   = blockIdx.x / NGT;       // batch group: 0..7
    const int ht   = blockIdx.x - bp * NGT;  // h-tile: 0..18
    const int rowbase = bp * 32;             // group's first batch row

    unsigned* cnt = bar + bp * 64;           // per-group barrier state
    unsigned* gen = cnt + 1;

    const int akb  = (lane >> 4) * 16;
    const int bkb  = (lane >> 4) * 16;
    const int gcol = gate * HT + ht * 16 + (lane & 15);
    const int lrow = bsel * 16 + (lane & 15);   // A row within the 32-row slab

    // ---- preload this wave's weight fragments; resident for all 1024 steps
    FragBF wih0[KC], whh0[KC], wih1[KC], whh1[KC];
    {
        const unsigned short* p0 = Wih0 + (size_t)gcol * Hp;
        const unsigned short* p1 = Whh0 + (size_t)gcol * Hp;
        const unsigned short* p2 = Wih1 + (size_t)gcol * Hp;
        const unsigned short* p3 = Whh1 + (size_t)gcol * Hp;
#pragma unroll
        for (int kc = 0; kc < KC; ++kc) {
            load_bfrag(wih0[kc], p0, kc * 32 + bkb);
            load_bfrag(whh0[kc], p1, kc * 32 + bkb);
            load_bfrag(wih1[kc], p2, kc * 32 + bkb);
            load_bfrag(whh1[kc], p3, kc * 32 + bkb);
        }
    }

    // ---- hoist biases for this thread's 2 cell-update elements
    float biasr[2][4][2];
#pragma unroll
    for (int ee = 0; ee < 2; ++ee) {
        const int hl   = (threadIdx.x + ee * NTHR) & 15;
        const int hcol = ht * 16 + hl;
#pragma unroll
        for (int g = 0; g < 4; ++g) {
            biasr[0][g][ee] = biasP0[g * HT + hcol];
            biasr[1][g][ee] = biasP1[g * HT + hcol];
        }
    }

    // zero c state (lives in LDS for the whole kernel)
    for (int i = threadIdx.x; i < 2 * 2 * 16 * 16; i += NTHR)
        (&c_s[0][0][0][0])[i] = 0.0f;
    __syncthreads();

    uint4* a4 = (uint4*)&a_s[0][0][0];            // 2560 uint4 units
    const unsigned short* ap1 = &a_s[0][lrow][0]; // this wave's A rows (LDS)
    const unsigned short* ap2 = &a_s[1][lrow][0];

    // Pull both A matrices' 32-row slabs from L2 straight into LDS with the
    // CDNA5 async path (tracked by ASYNCcnt; no VGPR round trip).
    auto stage_a = [&](const unsigned short* __restrict__ m0,
                       const unsigned short* __restrict__ m1) {
#pragma unroll
        for (int i = 0; i < 10; ++i) {
            const int u     = threadIdx.x + i * NTHR;        // 0..2559
            const int inner = threadIdx.x + (i % 5) * NTHR;  // 0..1279
            const unsigned short* src = (i < 5) ? m0 : m1;
            const int r  = inner / 40;                       // 40 uint4 / row
            const int c8 = inner - r * 40;
            const void* gaddr =
                src + (size_t)(rowbase + r) * Hp + (size_t)c8 * 8;
            const unsigned lds_off = (unsigned)(uintptr_t)(a4 + u);
            asm volatile("global_load_async_to_lds_b128 %0, %1, off"
                         :: "v"(lds_off), "v"(gaddr)
                         : "memory");
        }
        asm volatile("s_wait_asynccnt 0x0" ::: "memory");
    };

    // one 16x16 gate tile from the LDS A slab + register-resident weights
    auto gates_tile = [&](const FragBF* w1, const FragBF* w2) -> v8f {
        v8f acc = {};
#pragma unroll
        for (int kc = 0; kc < KC; ++kc) {
            FragBF a1, a2;
            a1.q[0] = *(const uint4*)(ap1 + kc * 32 + akb);
            a1.q[1] = *(const uint4*)(ap1 + kc * 32 + akb + 8);
            a2.q[0] = *(const uint4*)(ap2 + kc * 32 + akb);
            a2.q[1] = *(const uint4*)(ap2 + kc * 32 + akb + 8);
            acc = __builtin_amdgcn_wmma_f32_16x16x32_bf16(
                false, a1.v, false, w1[kc].v, (short)0, acc, false, false);
            acc = __builtin_amdgcn_wmma_f32_16x16x32_bf16(
                false, a2.v, false, w2[kc].v, (short)0, acc, false, false);
        }
        return acc;
    };

    auto store_gates = [&](v8f acc) {
        const int col = lane & 15;
        const int r0  = (lane >> 4) * 8;
#pragma unroll
        for (int r = 0; r < 8; ++r)
            gates_s[bsel][gate][r0 + r][col] = acc[r];
    };

    // block-local LSTM cell update over the block's 2x16x16 elements
    auto cell_update = [&](int layer, unsigned short* __restrict__ hdst,
                           unsigned short* __restrict__ hdst2,
                           float* __restrict__ odst, int t) {
#pragma unroll
        for (int ee = 0; ee < 2; ++ee) {
            const int e     = threadIdx.x + ee * NTHR;
            const int ebsel = e >> 8;
            const int rem   = e & 255;
            const int bl    = rem >> 4;
            const int hl    = rem & 15;
            const int hcol  = ht * 16 + hl;
            const int brow  = rowbase + ebsel * 16 + bl;
            const float gi = gates_s[ebsel][0][bl][hl] + biasr[layer][0][ee];
            const float gf = gates_s[ebsel][1][bl][hl] + biasr[layer][1][ee];
            const float gg = gates_s[ebsel][2][bl][hl] + biasr[layer][2][ee];
            const float go = gates_s[ebsel][3][bl][hl] + biasr[layer][3][ee];
            const float i_ = sigmoidf_(gi);
            const float f_ = sigmoidf_(gf);
            const float o_ = sigmoidf_(go);
            const float cn = f_ * c_s[layer][ebsel][bl][hl] + i_ * tanhf(gg);
            c_s[layer][ebsel][bl][hl] = cn;
            const float hn = o_ * tanhf(cn);
            const unsigned short hb = f2bf(hn);
            hdst[(size_t)brow * Hp + hcol] = hb;     // pad cols are harmless:
            if (hdst2)                               // W rows k>=300 are zero
                hdst2[(size_t)brow * Hp + hcol] = hb;
            if (odst && hcol < Hh)
                odst[(size_t)brow * Tt * Hh + (size_t)t * Hh + hcol] = hn;
        }
    };

    for (int t = 0; t < Tt; ++t) {
        // ---------------- layer 0: gates = x@Wih0 + h0@Whh0 ----------------
        stage_a(bx, bh0);
        __syncthreads();
        store_gates(gates_tile(wih0, whh0));
        __syncthreads();
        cell_update(0, bh0, nullptr, nullptr, t);   // publishes h0
        group_barrier(cnt, gen);                    // h0 visible in group

        // ---------------- layer 1: gates = h0@Wih1 + h1@Whh1 ---------------
        stage_a(bh0, bh1);
        __syncthreads();
        store_gates(gates_tile(wih1, whh1));
        __syncthreads();
        cell_update(1, bh1, bx, out, t);            // h1 -> bh1, bx, out
        group_barrier(cnt, gen);                    // h1/x visible in group
    }
}

// ----------------------------- host side ------------------------------------
extern "C" void kernel_launch(void* const* d_in, const int* in_sizes, int n_in,
                              void* d_out, int out_size, void* d_ws,
                              size_t ws_size, hipStream_t stream)
{
    (void)in_sizes; (void)n_in; (void)out_size; (void)ws_size;

    const float* tag  = (const float*)d_in[0];
    // d_in[1] = seq_len (scalar, fixed at 1024)
    const float* fcw  = (const float*)d_in[2];
    const float* fcb  = (const float*)d_in[3];
    const float* wih0 = (const float*)d_in[4];
    const float* whh0 = (const float*)d_in[5];
    const float* bih0 = (const float*)d_in[6];
    const float* bhh0 = (const float*)d_in[7];
    const float* wih1 = (const float*)d_in[8];
    const float* whh1 = (const float*)d_in[9];
    const float* bih1 = (const float*)d_in[10];
    const float* bhh1 = (const float*)d_in[11];
    float* out = (float*)d_out;

    char* base = (char*)d_ws;
    size_t off = 0;
    auto take = [&](size_t bytes) -> char* {
        char* p = base + off;
        off = (off + bytes + 255) & ~(size_t)255;
        return p;
    };

    unsigned* bar = (unsigned*)take(8 * 64 * sizeof(unsigned));
    unsigned short* bx  = (unsigned short*)take((size_t)Bb * Hp * 2);
    unsigned short* bh0 = (unsigned short*)take((size_t)Bb * Hp * 2);
    unsigned short* bh1 = (unsigned short*)take((size_t)Bb * Hp * 2);
    float* biasP0 = (float*)take((size_t)Gp * 4);
    float* biasP1 = (float*)take((size_t)Gp * 4);
    unsigned short* Wih0 = (unsigned short*)take((size_t)Gp * Hp * 2);
    unsigned short* Whh0 = (unsigned short*)take((size_t)Gp * Hp * 2);
    unsigned short* Wih1 = (unsigned short*)take((size_t)Gp * Hp * 2);
    unsigned short* Whh1 = (unsigned short*)take((size_t)Gp * Hp * 2);

    // 1) zero state / fold+pack biases / reset barriers (ws poisoned per call)
    {
        const int n = Bb * Hp;
        init_state_kernel<<<(n + NTHR - 1) / NTHR, NTHR, 0, stream>>>(
            bar, bx, bh0, bh1, bih0, bhh0, bih1, bhh1, biasP0, biasP1);
    }
    // 2) weights -> bf16, repacked to [Gp][Hp]
    {
        const int n = Gp * Hp;
        const int nb = (n + NTHR - 1) / NTHR;
        conv_w_kernel<<<nb, NTHR, 0, stream>>>(wih0, Wih0);
        conv_w_kernel<<<nb, NTHR, 0, stream>>>(whh0, Whh0);
        conv_w_kernel<<<nb, NTHR, 0, stream>>>(wih1, Wih1);
        conv_w_kernel<<<nb, NTHR, 0, stream>>>(whh1, Whh1);
    }
    // 3) x0 = tag @ fc_w^T + fc_b  -> bx (bf16)
    {
        const int n = Bb * Hh;
        fc_proj_kernel<<<(n + NTHR - 1) / NTHR, NTHR, 0, stream>>>(
            tag, fcw, fcb, bx);
    }
    // 4) persistent 1024-step LSTM (8 independent batch groups of 19 blocks)
    lstm_persist_kernel<<<NWG, NTHR, 0, stream>>>(
        bx, bh0, bh1, Wih0, Whh0, Wih1, Whh1, biasP0, biasP1, out, bar);
}
